// SGLossOpt2_10084583211023
// MI455X (gfx1250) — matrix operand order repaired and checked by reading, not compiled, not run on hardware
//
#include <hip/hip_runtime.h>
#include <hip/hip_bf16.h>

#define BATCH 4096
#define HDIM  1024
#define INV_TEMP 20.0f
#define NCB   16                    // column blocks for the lse partial split
#define COLS_PER_CB (BATCH / NCB)   // 256

typedef __attribute__((ext_vector_type(16))) __bf16 v16bf;
typedef __attribute__((ext_vector_type(8)))  float  v8f;

struct BFrag { union { v16bf v; uint4 q[2]; }; };

__device__ __forceinline__ unsigned short f2bf(float f) {
    unsigned int u = __float_as_uint(f);
    unsigned int r = u + 0x7FFFu + ((u >> 16) & 1u);   // round-to-nearest-even
    return (unsigned short)(r >> 16);
}

// A fragment (16x32 bf16), ISA layout:
//  lanes 0-15 : V0-3 <- K 0..7  (bytes  0..15), V4-7 <- K16..23 (bytes 32..47)
//  lanes16-31 : V0-3 <- K 8..15 (bytes 16..31), V4-7 <- K24..31 (bytes 48..63)
__device__ __forceinline__ void load_fragA(BFrag& d, const unsigned short* arow,
                                           int k0, int half) {
    const uint4* ap = (const uint4*)(arow + k0);
    d.q[0] = ap[half];
    d.q[1] = ap[2 + half];
}

// B fragment (32x16 bf16; col j = hidden row j, K contiguous):
//  lanes 0-15 : V0-3 <- K 0..7  (bytes  0..15), V4-7 <- K16..23 (bytes 32..47)
//  lanes16-31 : V0-3 <- K 8..15 (bytes 16..31), V4-7 <- K24..31 (bytes 48..63)
__device__ __forceinline__ void load_fragB(BFrag& d, const unsigned short* brow,
                                           int k0, int half) {
    const uint4* bp = (const uint4*)(brow + k0);
    d.q[0] = bp[half];
    d.q[1] = bp[2 + half];
}

// ---------------------------------------------------------------------------
// Kernel 1: per-row L2 normalization (f32), bf16 export, exact f32 diag.
// ---------------------------------------------------------------------------
__global__ __launch_bounds__(256)
void norm_diag_kernel(const float* __restrict__ cls,
                      const float* __restrict__ hid,
                      unsigned short* __restrict__ a16,
                      unsigned short* __restrict__ b16,
                      float* __restrict__ diag)
{
    const int r = blockIdx.x;
    const int t = threadIdx.x;                 // 0..255
    const float4 a4 = ((const float4*)(cls + (size_t)r * HDIM))[t];
    const float4 b4 = ((const float4*)(hid + (size_t)r * HDIM))[t];

    float ssa = a4.x * a4.x + a4.y * a4.y + a4.z * a4.z + a4.w * a4.w;
    float ssb = b4.x * b4.x + b4.y * b4.y + b4.z * b4.z + b4.w * b4.w;

    __shared__ float sA[8], sB[8], sD[8];
    #pragma unroll
    for (int m = 16; m >= 1; m >>= 1) {
        ssa += __shfl_xor(ssa, m, 32);
        ssb += __shfl_xor(ssb, m, 32);
    }
    if ((t & 31) == 0) { sA[t >> 5] = ssa; sB[t >> 5] = ssb; }
    __syncthreads();
    float na = 0.f, nb = 0.f;
    #pragma unroll
    for (int i = 0; i < 8; i++) { na += sA[i]; nb += sB[i]; }

    const float ia = 1.0f / fmaxf(sqrtf(na), 1e-8f);
    const float ib = 1.0f / fmaxf(sqrtf(nb), 1e-8f);

    const float anx = a4.x * ia, any_ = a4.y * ia, anz = a4.z * ia, anw = a4.w * ia;
    const float bnx = b4.x * ib, bny  = b4.y * ib, bnz = b4.z * ib, bnw = b4.w * ib;

    float d = anx * bnx + any_ * bny + anz * bnz + anw * bnw;   // exact f32 diag
    #pragma unroll
    for (int m = 16; m >= 1; m >>= 1) d += __shfl_xor(d, m, 32);
    if ((t & 31) == 0) sD[t >> 5] = d;
    __syncthreads();
    if (t == 0) {
        float dd = 0.f;
        #pragma unroll
        for (int i = 0; i < 8; i++) dd += sD[i];
        diag[r] = dd * INV_TEMP;
    }

    uint2 pa, pb;
    pa.x = (unsigned int)f2bf(anx) | ((unsigned int)f2bf(any_) << 16);
    pa.y = (unsigned int)f2bf(anz) | ((unsigned int)f2bf(anw) << 16);
    pb.x = (unsigned int)f2bf(bnx) | ((unsigned int)f2bf(bny) << 16);
    pb.y = (unsigned int)f2bf(bnz) | ((unsigned int)f2bf(bnw) << 16);
    ((uint2*)(a16 + (size_t)r * HDIM))[t] = pa;
    ((uint2*)(b16 + (size_t)r * HDIM))[t] = pb;
}

// ---------------------------------------------------------------------------
// Kernel 2: streaming bf16 WMMA GEMM + partial sum-of-exp per row.
// Grid: (BATCH/256, NCB). 8 waves/WG; wave w owns rows [bx*256 + 32w, +32).
// Wave tile: 32 rows x 64 cols = 2x4 C frags. Per 32-wide K step: one batched
// clause of 12 b128 loads (2 A frags + 4 B frags), then 8 back-to-back WMMAs.
// Single-buffered on purpose: keeps VGPRs < 256 (no spills / no MSB mode);
// inter-wave overlap hides the per-step load wait.
// ---------------------------------------------------------------------------
__global__ __launch_bounds__(256)
void gemm_lse_partial(const unsigned short* __restrict__ A,
                      const unsigned short* __restrict__ Bm,
                      float* __restrict__ partial)
{
    const int lane = threadIdx.x & 31;
    const int wave = threadIdx.x >> 5;
    const int half = lane >> 4;      // 0: lanes 0-15, 1: lanes 16-31
    const int l15  = lane & 15;
    const int row_base = blockIdx.x * 256 + wave * 32;
    const int col_lo   = blockIdx.y * COLS_PER_CB;

    const unsigned short* arow0 = A + (size_t)(row_base + l15) * HDIM;
    const unsigned short* arow1 = arow0 + 16 * HDIM;

    float acc[16];
    #pragma unroll
    for (int i = 0; i < 16; i++) acc[i] = 0.f;

    for (int jt = col_lo; jt < col_lo + COLS_PER_CB; jt += 64) {
        const unsigned short* brow0 = Bm + (size_t)(jt + l15) * HDIM;

        v8f c[2][4];
        #pragma unroll
        for (int m = 0; m < 2; m++)
            #pragma unroll
            for (int n = 0; n < 4; n++) c[m][n] = (v8f)(0.0f);

        for (int k0 = 0; k0 < HDIM; k0 += 32) {
            BFrag a0, a1, b0, b1, b2, b3;
            // batch all 12 B128 loads for this K step
            load_fragA(a0, arow0, k0, half);
            load_fragA(a1, arow1, k0, half);
            load_fragB(b0, brow0 + 0 * 16 * HDIM, k0, half);
            load_fragB(b1, brow0 + 1 * 16 * HDIM, k0, half);
            load_fragB(b2, brow0 + 2 * 16 * HDIM, k0, half);
            load_fragB(b3, brow0 + 3 * 16 * HDIM, k0, half);

            // 8 back-to-back WMMAs; each B fragment reused by both A frags
            c[0][0] = __builtin_amdgcn_wmma_f32_16x16x32_bf16(
                          false, a0.v, false, b0.v, (short)0, c[0][0], false, false);
            c[1][0] = __builtin_amdgcn_wmma_f32_16x16x32_bf16(
                          false, a1.v, false, b0.v, (short)0, c[1][0], false, false);
            c[0][1] = __builtin_amdgcn_wmma_f32_16x16x32_bf16(
                          false, a0.v, false, b1.v, (short)0, c[0][1], false, false);
            c[1][1] = __builtin_amdgcn_wmma_f32_16x16x32_bf16(
                          false, a1.v, false, b1.v, (short)0, c[1][1], false, false);
            c[0][2] = __builtin_amdgcn_wmma_f32_16x16x32_bf16(
                          false, a0.v, false, b2.v, (short)0, c[0][2], false, false);
            c[1][2] = __builtin_amdgcn_wmma_f32_16x16x32_bf16(
                          false, a1.v, false, b2.v, (short)0, c[1][2], false, false);
            c[0][3] = __builtin_amdgcn_wmma_f32_16x16x32_bf16(
                          false, a0.v, false, b3.v, (short)0, c[0][3], false, false);
            c[1][3] = __builtin_amdgcn_wmma_f32_16x16x32_bf16(
                          false, a1.v, false, b3.v, (short)0, c[1][3], false, false);
        }

        // epilogue: sum exp(logit) into per-row accumulators.
        // C layout: elem r of v8f = VGPR r; lanes 0-15 -> row m*16+r, lanes16-31 -> row m*16+8+r.
        #pragma unroll
        for (int m = 0; m < 2; m++)
            #pragma unroll
            for (int n = 0; n < 4; n++)
                #pragma unroll
                for (int r = 0; r < 8; r++)
                    acc[m * 8 + r] += __expf(c[m][n][r] * INV_TEMP);
    }

    // reduce the 16 columns held across lanes of each half-wave
    #pragma unroll
    for (int m = 1; m < 16; m <<= 1) {
        #pragma unroll
        for (int i = 0; i < 16; i++) acc[i] += __shfl_xor(acc[i], m, 32);
    }
    if (l15 == 0) {
        float* p = partial + (size_t)blockIdx.y * BATCH + row_base;
        #pragma unroll
        for (int m = 0; m < 2; m++)
            #pragma unroll
            for (int r = 0; r < 8; r++)
                p[m * 16 + half * 8 + r] = acc[m * 8 + r];
    }
}

// ---------------------------------------------------------------------------
// Kernel 3: combine partials -> mean(log(sumexp) - diag)
// ---------------------------------------------------------------------------
__global__ __launch_bounds__(256)
void finalize_kernel(const float* __restrict__ partial,
                     const float* __restrict__ diag,
                     float* __restrict__ out)
{
    const int t = threadIdx.x;
    float local = 0.f;
    for (int i = t; i < BATCH; i += 256) {
        float s = 0.f;
        #pragma unroll
        for (int cb = 0; cb < NCB; cb++) s += partial[(size_t)cb * BATCH + i];
        local += logf(s) - diag[i];
    }
    __shared__ float sm[8];
    #pragma unroll
    for (int m = 16; m >= 1; m >>= 1) local += __shfl_xor(local, m, 32);
    if ((t & 31) == 0) sm[t >> 5] = local;
    __syncthreads();
    if (t == 0) {
        float tot = 0.f;
        #pragma unroll
        for (int i = 0; i < 8; i++) tot += sm[i];
        out[0] = tot / (float)BATCH;
    }
}

// ---------------------------------------------------------------------------
extern "C" void kernel_launch(void* const* d_in, const int* in_sizes, int n_in,
                              void* d_out, int out_size, void* d_ws, size_t ws_size,
                              hipStream_t stream) {
    const float* cls = (const float*)d_in[0];
    const float* hid = (const float*)d_in[1];

    char* ws = (char*)d_ws;
    const size_t bf_bytes = (size_t)BATCH * HDIM * sizeof(unsigned short); // 8 MB
    unsigned short* a16  = (unsigned short*)(ws);
    unsigned short* b16  = (unsigned short*)(ws + bf_bytes);
    float*          diag = (float*)(ws + 2 * bf_bytes);
    float*          part = (float*)(ws + 2 * bf_bytes + (size_t)BATCH * sizeof(float));

    norm_diag_kernel<<<BATCH, 256, 0, stream>>>(cls, hid, a16, b16, diag);

    dim3 grid(BATCH / 256, NCB);
    gemm_lse_partial<<<grid, 256, 0, stream>>>(a16, b16, part);

    finalize_kernel<<<1, 256, 0, stream>>>(part, diag, (float*)d_out);
}
